// NyonicAttention_79242146611331
// MI455X (gfx1250) — matrix operand order repaired and checked by reading, not compiled, not run on hardware
//
#include <hip/hip_runtime.h>
#include <stdint.h>

// ---------------------------------------------------------------------------
// NyonicAttention for MI455X (gfx1250): bf16 WMMA everywhere matrix-shaped.
// B=2, T=2048, D=2048, NH=16, HD=128. ~206 GFLOP vs ~170MB traffic at
// 23.3 TB/s => compute bound => v_wmma_f32_16x16x32_bf16 with f32 accum.
// Tile staging uses the Tensor Data Mover (tensor_load_to_lds + TENSORcnt)
// with hardware row padding; falls back to manual staging if the builtin
// is unavailable.
// ---------------------------------------------------------------------------

typedef __attribute__((ext_vector_type(16))) __bf16 v16bf;
typedef __attribute__((ext_vector_type(8)))  float  v8f;
typedef __attribute__((ext_vector_type(4)))  unsigned int v4u;
typedef __attribute__((ext_vector_type(4)))  unsigned int u32x4;
typedef __attribute__((ext_vector_type(8)))  int i32x8;
typedef __attribute__((ext_vector_type(4)))  int i32x4;

#define B_  2
#define T_  2048
#define D_  2048
#define NH_ 16
#define HD_ 128

#if __has_builtin(__builtin_amdgcn_tensor_load_to_lds) && \
    __has_builtin(__builtin_amdgcn_s_wait_tensorcnt)
#define HAVE_TDM 1
#else
#define HAVE_TDM 0
#endif

union FragU { v16bf v; v4u u[2]; __bf16 h[16]; };

__device__ __forceinline__ v16bf frag2(const void* p0, const void* p1) {
  FragU f;
  f.u[0] = *(const v4u*)p0;
  f.u[1] = *(const v4u*)p1;
  return f.v;
}

#if HAVE_TDM
// 2-D tile DMA: global (row-major, d0 innermost) -> LDS with optional row pad.
// Descriptor per CDNA5 ISA 8.3/8.4: group0 = {flags, lds_addr, gaddr lo/hi|type},
// group1 = {mask/size/pad, tensor dims, tile dims, strides}. data_size=2B.
// This toolchain exposes the 6-arg builtin: (g0, g1, g2, g3, g4, cpol) where
// the trailing int32x8 group is unused for a plain 2-D descriptor.
__device__ __forceinline__ void tdm_load_2d(const void* gaddr, unsigned lds_addr,
                                            unsigned tensor_d0, unsigned tensor_d1,
                                            unsigned d0_stride,
                                            unsigned tile_d0, unsigned tile_d1,
                                            unsigned pad_interval, unsigned pad_amount) {
  unsigned long long ga = (unsigned long long)(uintptr_t)gaddr;
  u32x4 g0;
  g0.x = 1u;                                   // count=1, user descriptor
  g0.y = lds_addr;                             // LDS byte address
  g0.z = (unsigned)ga;                         // global addr [31:0]
  g0.w = (unsigned)(ga >> 32) | (2u << 30);    // global addr [56:32] | type=2
  i32x8 g1;
  g1[0] = (int)((1u << 16)                     // data_size = 2 bytes
                | (1u << 20)                   // pad_enable
                | (pad_interval << 22)         // pad every 2^(v+1) DWORDs
                | (pad_amount << 25));         // pad (v+1) DWORDs
  g1[1] = (int)(tensor_d0 << 16);              // tensor_dim0 [79:48]
  g1[2] = (int)((tensor_d0 >> 16) | (tensor_d1 << 16));
  g1[3] = (int)((tensor_d1 >> 16) | (tile_d0 << 16));
  g1[4] = (int)(tile_d1 & 0xFFFFu);            // tile_dim1 (tile_dim2 = 0)
  g1[5] = (int)d0_stride;                      // tensor_dim0_stride [31:0]
  g1[6] = 0;                                   // stride hi, dim1_stride lo
  g1[7] = 0;
  i32x4 z4 = {0, 0, 0, 0};                     // groups 2/3 unused (2-D tensor)
  i32x8 z8 = {0, 0, 0, 0, 0, 0, 0, 0};         // trailing group unused
  __builtin_amdgcn_tensor_load_to_lds(g0, g1, z4, z4, z8, 0);
}
#endif

// ---------------------------------------------------------------------------
// fp32 -> bf16 conversion (4-wide, grid-stride)
// ---------------------------------------------------------------------------
struct bf4 { __bf16 x, y, z, w; };

__global__ void cvt_f32_bf16(const float* __restrict__ src,
                             __bf16* __restrict__ dst, long n4) {
  long i = (long)blockIdx.x * blockDim.x + threadIdx.x;
  long stride = (long)gridDim.x * blockDim.x;
  for (; i < n4; i += stride) {
    float4 f = ((const float4*)src)[i];
    bf4 o;
    o.x = (__bf16)f.x; o.y = (__bf16)f.y;
    o.z = (__bf16)f.z; o.w = (__bf16)f.w;
    ((bf4*)dst)[i] = o;
  }
}

// ---------------------------------------------------------------------------
// GEMM: C[m,n] = sum_k A[m,k] * W[n,k]   (A:[M,K] bf16, W:[N,K] bf16)
// 128x128 block tile, K-step 32, 8 waves each computing 32x64.
// Double-buffered LDS, 80B row pitch (64B data + 16B TDM pad) for 16B-aligned
// b128 fragment loads with rows spread across banks. Tiles staged by the
// Tensor Data Mover (wave0 -> A, wave1 -> W), overlapped with WMMA.
// ---------------------------------------------------------------------------
template<bool F32OUT>
__global__ __launch_bounds__(256)
void gemm_bt(const __bf16* __restrict__ A, const __bf16* __restrict__ W,
             float* __restrict__ Cf, __bf16* __restrict__ Cb,
             int M, int N, int K) {
  constexpr int PITCH = 80;                 // bytes per 32-bf16 LDS row
  constexpr int TSZ   = 128 * PITCH;        // one operand tile: 10240 B
  __shared__ char lds[2 * 2 * TSZ];         // double buffered A+W: 40 KB

  const int tid  = threadIdx.x;
  const int lane = tid & 31;
  const int w    = tid >> 5;
  const int wm   = w >> 1;                  // 0..3  (row group of 32)
  const int wn   = w & 1;                   // 0..1  (col group of 64)
  const int l15  = lane & 15;
  const int half = lane >> 4;

  const int m0 = blockIdx.y * 128;
  const int n0 = blockIdx.x * 128;

  v8f acc[2][4] = {};

#if HAVE_TDM
  const unsigned lds0 = (unsigned)(uintptr_t)(void*)lds;
  // pad_interval=3: pad after every 16 DWORDs (64B); pad_amount=3: 4 DWORDs (16B)
  auto stage = [&](int bufsel, int k0) {
    if (w == 0)
      tdm_load_2d(A + (size_t)m0 * K + k0, lds0 + bufsel * (2 * TSZ),
                  (unsigned)K, (unsigned)M, (unsigned)K, 32u, 128u, 3u, 3u);
    else if (w == 1)
      tdm_load_2d(W + (size_t)n0 * K + k0, lds0 + bufsel * (2 * TSZ) + TSZ,
                  (unsigned)K, (unsigned)N, (unsigned)K, 32u, 128u, 3u, 3u);
  };
  stage(0, 0);
  if (w < 2) __builtin_amdgcn_s_wait_tensorcnt(0);
  __syncthreads();
#else
  v4u sa[2], sb[2];
  auto stage_regs = [&](int k0) {
#pragma unroll
    for (int it = 0; it < 2; ++it) {
      int idx = tid + it * 256;
      int r = idx >> 2, c = idx & 3;
      sa[it] = *(const v4u*)(A + (size_t)(m0 + r) * K + k0 + c * 8);
      sb[it] = *(const v4u*)(W + (size_t)(n0 + r) * K + k0 + c * 8);
    }
  };
  auto stage_lds = [&](char* buf) {
#pragma unroll
    for (int it = 0; it < 2; ++it) {
      int idx = tid + it * 256;
      int r = idx >> 2, c = idx & 3;
      *(v4u*)(buf + r * PITCH + c * 16)       = sa[it];
      *(v4u*)(buf + TSZ + r * PITCH + c * 16) = sb[it];
    }
  };
  stage_regs(0);
  stage_lds(lds);
  __syncthreads();
#endif

  const int nk = K >> 5;
  for (int kt = 0; kt < nk; ++kt) {
    char* cur = lds + (kt & 1) * (2 * TSZ);
    const bool more = (kt + 1) < nk;
#if HAVE_TDM
    if (more) stage((kt + 1) & 1, (kt + 1) * 32);   // DMA next tile during WMMA
#else
    if (more) stage_regs((kt + 1) * 32);
#endif

    v16bf af[2], bfr[4];
#pragma unroll
    for (int mi = 0; mi < 2; ++mi) {
      // A layout: lane holds row (wm*32+mi*16+l15); K chunks half*8 and 16+half*8
      char* p = cur + (wm * 32 + mi * 16 + l15) * PITCH + half * 16;
      af[mi] = frag2(p, p + 32);
    }
#pragma unroll
    for (int ni = 0; ni < 4; ++ni) {
      // B layout: lane holds col (wn*64+ni*16+l15); K chunks half*8 and 16+half*8
      char* p = cur + TSZ + (wn * 64 + ni * 16 + l15) * PITCH + half * 16;
      bfr[ni] = frag2(p, p + 32);
    }
#pragma unroll
    for (int mi = 0; mi < 2; ++mi)
#pragma unroll
      for (int ni = 0; ni < 4; ++ni)
        acc[mi][ni] = __builtin_amdgcn_wmma_f32_16x16x32_bf16(
            false, af[mi], false, bfr[ni], (short)0, acc[mi][ni], false, false);

#if HAVE_TDM
    if (more && w < 2) __builtin_amdgcn_s_wait_tensorcnt(0);
#else
    if (more) stage_lds(lds + ((kt + 1) & 1) * (2 * TSZ));
#endif
    __syncthreads();
  }

  // C layout: lane half picks M 0-7 vs 8-15; lane&15 is N
  const int rbase = m0 + wm * 32 + half * 8;
  const int cbase = n0 + wn * 64 + l15;
#pragma unroll
  for (int mi = 0; mi < 2; ++mi)
#pragma unroll
    for (int ni = 0; ni < 4; ++ni)
#pragma unroll
      for (int r = 0; r < 8; ++r) {
        size_t row = (size_t)(rbase + mi * 16 + r);
        size_t col = (size_t)(cbase + ni * 16);
        float v = acc[mi][ni][r];
        if (F32OUT) Cf[row * N + col] = v;
        else        Cb[row * N + col] = (__bf16)v;
      }
}

// ---------------------------------------------------------------------------
// Per-head LayerNorm + RoPE. One block per token, one wave per head
// (HD=128 = 4 values/lane on wave32). RoPE pair (i, i+64) is same lane.
// Writes q,k,v bf16 in head-major layout [B, NH, T, HD].
// ---------------------------------------------------------------------------
__global__ __launch_bounds__(256)
void ln_rope(const __bf16* __restrict__ qkv,
             const float* __restrict__ qsc, const float* __restrict__ qbi,
             const float* __restrict__ ksc, const float* __restrict__ kbi,
             __bf16* __restrict__ qo, __bf16* __restrict__ ko,
             __bf16* __restrict__ vo) {
  const int tok  = blockIdx.x;
  const int b    = tok / T_;
  const int t    = tok % T_;
  const int w    = threadIdx.x >> 5;
  const int lane = threadIdx.x & 31;
  const float LN1E4_64 = 9.210340371976184f / 64.0f;   // ln(10000)/64

  for (int h = w; h < NH_; h += 8) {
    const size_t base = (size_t)tok * (3 * D_) + (size_t)h * HD_;
    float qv[4], kv[4];
#pragma unroll
    for (int j = 0; j < 4; ++j) {
      int i = lane + j * 32;
      qv[j] = (float)qkv[base + i];
      kv[j] = (float)qkv[base + D_ + i];
    }
    float qs = qv[0] + qv[1] + qv[2] + qv[3];
    float ks = kv[0] + kv[1] + kv[2] + kv[3];
    float qq = qv[0]*qv[0] + qv[1]*qv[1] + qv[2]*qv[2] + qv[3]*qv[3];
    float kk = kv[0]*kv[0] + kv[1]*kv[1] + kv[2]*kv[2] + kv[3]*kv[3];
#pragma unroll
    for (int off = 1; off < 32; off <<= 1) {
      qs += __shfl_xor(qs, off, 32);
      ks += __shfl_xor(ks, off, 32);
      qq += __shfl_xor(qq, off, 32);
      kk += __shfl_xor(kk, off, 32);
    }
    const float qmu = qs * (1.0f / HD_), kmu = ks * (1.0f / HD_);
    const float qrs = rsqrtf(qq * (1.0f / HD_) - qmu * qmu + 1e-5f);
    const float krs = rsqrtf(kk * (1.0f / HD_) - kmu * kmu + 1e-5f);

    float nq[4], nk_[4];
#pragma unroll
    for (int j = 0; j < 4; ++j) {
      int i = lane + j * 32;
      nq[j]  = (qv[j] - qmu) * qrs * qsc[i] + qbi[i];
      nk_[j] = (kv[j] - kmu) * krs * ksc[i] + kbi[i];
    }
    const size_t ob = ((size_t)(b * NH_ + h) * T_ + t) * HD_;
#pragma unroll
    for (int j = 0; j < 4; ++j) {
      int i  = lane + j * 32;
      int jj = i & 63;
      float sn, cs;
      __sincosf((float)t * __expf(-(float)jj * LN1E4_64), &sn, &cs);
      float oq, okv;
      if (j < 2) { oq = nq[j] * cs - nq[j + 2] * sn;
                   okv = nk_[j] * cs - nk_[j + 2] * sn; }
      else       { oq = nq[j] * cs + nq[j - 2] * sn;
                   okv = nk_[j] * cs + nk_[j - 2] * sn; }
      qo[ob + i] = (__bf16)oq;
      ko[ob + i] = (__bf16)okv;
      vo[ob + i] = qkv[base + 2 * D_ + i];   // v: straight copy to bf16 layout
    }
  }
}

// ---------------------------------------------------------------------------
// Causal flash attention. Block = (b, h, 128-row q-tile), 8 waves of 16 rows.
// KV tiles of 64. K staged row-major in LDS via TDM (row pad in hardware),
// V staged transposed (register transpose) for the B operand of P@V.
// Probabilities round-trip through per-wave LDS to convert C-register layout
// into A-fragment layout. All matmuls bf16 WMMA, f32 accumulation.
// ---------------------------------------------------------------------------
__global__ __launch_bounds__(256)
void flash_attn(const __bf16* __restrict__ Q, const __bf16* __restrict__ Kk,
                const __bf16* __restrict__ V, __bf16* __restrict__ O) {
  constexpr int PK = 272;   // K tile: 64 rows x (128 bf16 + 16B pad)
  constexpr int PV = 144;   // VT tile: 128 hd rows x (64 bf16 + 16B pad)
  constexpr int PP = 144;   // P tile: per wave 16 rows x (64 bf16 + 16B pad)
  __shared__ char sK[64 * PK];        // 17408 B
  __shared__ char sVT[128 * PV];      // 18432 B
  __shared__ char sP[8 * 16 * PP];    // 18432 B   total 54272 B

  const int qt   = blockIdx.x;
  const int h    = blockIdx.y;
  const int b    = blockIdx.z;
  const int tid  = threadIdx.x;
  const int w    = tid >> 5;
  const int lane = tid & 31;
  const int l15  = lane & 15;
  const int half = lane >> 4;
  const int q0   = qt * 128;

  const __bf16* qh = Q  + (size_t)(b * NH_ + h) * T_ * HD_;
  const __bf16* kh = Kk + (size_t)(b * NH_ + h) * T_ * HD_;
  const __bf16* vh = V  + (size_t)(b * NH_ + h) * T_ * HD_;

  // Q fragments for this wave's 16 rows, 4 K-chunks of 32 over HD=128
  v16bf qf[4];
  {
    const __bf16* qr = qh + (size_t)(q0 + w * 16 + l15) * HD_;
#pragma unroll
    for (int kc = 0; kc < 4; ++kc) {
      const __bf16* p = qr + kc * 32 + half * 8;
      qf[kc] = frag2(p, p + 16);
    }
  }

  v8f oacc[8] = {};
  float rmax[8], rsum[8];
#pragma unroll
  for (int r = 0; r < 8; ++r) { rmax[r] = -__builtin_inff(); rsum[r] = 0.f; }
  const float scale = 0.08838834764831845f;   // 1/sqrt(128)
  char* Pw = sP + w * 16 * PP;
#if HAVE_TDM
  const unsigned sK0 = (unsigned)(uintptr_t)(void*)sK;
#endif

  const int nt = 2 * (qt + 1);                // causal: kv0 <= q0+64
  for (int kt = 0; kt < nt; ++kt) {
    const int kv0 = kt * 64;

    // ---- stage K tile (TDM DMA) and transposed V tile into LDS -----------
#if HAVE_TDM
    // 128x64 tile, 256B rows padded by 16B: pad_interval=5 (64 DW), amount=3
    if (w == 0)
      tdm_load_2d(kh + (size_t)kv0 * HD_, sK0,
                  (unsigned)HD_, (unsigned)T_, (unsigned)HD_, 128u, 64u, 5u, 3u);
#endif
#pragma unroll
    for (int it = 0; it < 4; ++it) {
      int idx = tid + it * 256;               // 64 rows x 16 chunks
      int r = idx >> 4, c = idx & 15;
#if !HAVE_TDM
      v4u kch = *(const v4u*)(kh + (size_t)(kv0 + r) * HD_ + c * 8);
      *(v4u*)(sK + r * PK + c * 16) = kch;
#endif
      FragU vu; vu.u[0] = *(const v4u*)(vh + (size_t)(kv0 + r) * HD_ + c * 8);
#pragma unroll
      for (int e = 0; e < 8; ++e)
        *(__bf16*)(sVT + (c * 8 + e) * PV + r * 2) = vu.h[e];
    }
#if HAVE_TDM
    if (w == 0) __builtin_amdgcn_s_wait_tensorcnt(0);
#endif
    __syncthreads();

    // ---- S = Q @ K^T (batch fragment loads, then back-to-back WMMA) ------
    v8f sacc[4];
#pragma unroll
    for (int nf = 0; nf < 4; ++nf) {
      v16bf bk[4];
#pragma unroll
      for (int kc = 0; kc < 4; ++kc) {
        char* p = sK + (nf * 16 + l15) * PK + (kc * 32 + half * 8) * 2;
        bk[kc] = frag2(p, p + 32);
      }
      v8f a = {};
#pragma unroll
      for (int kc = 0; kc < 4; ++kc)
        a = __builtin_amdgcn_wmma_f32_16x16x32_bf16(
            false, qf[kc], false, bk[kc], (short)0, a, false, false);
      sacc[nf] = a;
    }

    // ---- scale + causal mask ---------------------------------------------
    const bool diag = (kv0 + 63 > q0);
#pragma unroll
    for (int nf = 0; nf < 4; ++nf)
#pragma unroll
      for (int r = 0; r < 8; ++r) {
        float s = sacc[nf][r] * scale;
        if (diag) {
          int colg = kv0 + nf * 16 + l15;
          int rowg = q0 + w * 16 + half * 8 + r;
          if (colg > rowg) s = -__builtin_inff();
        }
        sacc[nf][r] = s;
      }

    // ---- online softmax: row max over tile (16-lane half reduction) ------
    float corr[8];
#pragma unroll
    for (int r = 0; r < 8; ++r) {
      float m = fmaxf(fmaxf(sacc[0][r], sacc[1][r]),
                      fmaxf(sacc[2][r], sacc[3][r]));
#pragma unroll
      for (int off = 1; off < 16; off <<= 1) m = fmaxf(m, __shfl_xor(m, off, 32));
      float mn = fmaxf(rmax[r], m);
      corr[r] = __expf(rmax[r] - mn);
      rmax[r] = mn;
    }

    // ---- p = exp(s-m): accumulate row sums, write bf16 P to per-wave LDS -
    float ts[8] = {};
#pragma unroll
    for (int nf = 0; nf < 4; ++nf)
#pragma unroll
      for (int r = 0; r < 8; ++r) {
        float p = __expf(sacc[nf][r] - rmax[r]);
        ts[r] += p;
        *(__bf16*)(Pw + (half * 8 + r) * PP + (nf * 16 + l15) * 2) = (__bf16)p;
      }
#pragma unroll
    for (int r = 0; r < 8; ++r) {
      float s = ts[r];
#pragma unroll
      for (int off = 1; off < 16; off <<= 1) s += __shfl_xor(s, off, 32);
      rsum[r] = rsum[r] * corr[r] + s;
    }
#pragma unroll
    for (int j = 0; j < 8; ++j)
#pragma unroll
      for (int r = 0; r < 8; ++r) oacc[j][r] *= corr[r];

    // ---- O += P @ V (A from P LDS, B from VT LDS; batched frag loads) ----
#pragma unroll
    for (int kc = 0; kc < 2; ++kc) {
      char* pa = Pw + l15 * PP + (kc * 32 + half * 8) * 2;
      v16bf af = frag2(pa, pa + 32);
#pragma unroll
      for (int jb = 0; jb < 2; ++jb) {
        v16bf bv[4];
#pragma unroll
        for (int jj = 0; jj < 4; ++jj) {
          int j = jb * 4 + jj;
          char* pb = sVT + (j * 16 + l15) * PV + (kc * 32 + half * 8) * 2;
          bv[jj] = frag2(pb, pb + 32);
        }
#pragma unroll
        for (int jj = 0; jj < 4; ++jj) {
          int j = jb * 4 + jj;
          oacc[j] = __builtin_amdgcn_wmma_f32_16x16x32_bf16(
              false, af, false, bv[jj], (short)0, oacc[j], false, false);
        }
      }
    }
    __syncthreads();
  }

  // ---- epilogue: O / rowsum -> attn bf16 [B, T, D] -----------------------
#pragma unroll
  for (int j = 0; j < 8; ++j)
#pragma unroll
    for (int r = 0; r < 8; ++r) {
      size_t row = (size_t)(q0 + w * 16 + half * 8 + r);
      size_t col = (size_t)(h * HD_ + j * 16 + l15);
      O[((size_t)b * T_ + row) * D_ + col] = (__bf16)(oacc[j][r] / rsum[r]);
    }
}

// ---------------------------------------------------------------------------
// Host-side launch
// ---------------------------------------------------------------------------
extern "C" void kernel_launch(void* const* d_in, const int* in_sizes, int n_in,
                              void* d_out, int out_size, void* d_ws, size_t ws_size,
                              hipStream_t stream) {
  (void)in_sizes; (void)n_in; (void)out_size; (void)ws_size;
  const float* x   = (const float*)d_in[0];
  const float* wi  = (const float*)d_in[1];
  const float* wo  = (const float*)d_in[2];
  const float* qsc = (const float*)d_in[3];
  const float* qbi = (const float*)d_in[4];
  const float* ksc = (const float*)d_in[5];
  const float* kbi = (const float*)d_in[6];
  float* out = (float*)d_out;

  char* ws = (char*)d_ws;
  size_t off = 0;
  auto carve = [&](size_t bytes) {
    char* p = ws + off;
    off += (bytes + 255) & ~(size_t)255;
    return p;
  };
  const size_t nX  = (size_t)B_ * T_ * D_;        // 8.4M
  const size_t nWi = (size_t)3 * D_ * D_;         // 12.6M
  const size_t nWo = (size_t)D_ * D_;             // 4.2M
  __bf16* xb   = (__bf16*)carve(nX  * 2);
  __bf16* wib  = (__bf16*)carve(nWi * 2);
  __bf16* wob  = (__bf16*)carve(nWo * 2);
  __bf16* qkvb = (__bf16*)carve((size_t)B_ * T_ * 3 * D_ * 2);
  __bf16* qo   = (__bf16*)carve(nX * 2);
  __bf16* ko   = (__bf16*)carve(nX * 2);
  __bf16* vo   = (__bf16*)carve(nX * 2);
  __bf16* ab   = (__bf16*)carve(nX * 2);

  // 1) fp32 -> bf16
  cvt_f32_bf16<<<2048, 256, 0, stream>>>(x,  xb,  (long)(nX  / 4));
  cvt_f32_bf16<<<2048, 256, 0, stream>>>(wi, wib, (long)(nWi / 4));
  cvt_f32_bf16<<<2048, 256, 0, stream>>>(wo, wob, (long)(nWo / 4));

  // 2) qkv = x @ Win^T : [4096,2048] x [6144,2048]^T -> bf16
  {
    dim3 g(6144 / 128, 4096 / 128);
    gemm_bt<false><<<g, 256, 0, stream>>>(xb, wib, nullptr, qkvb,
                                          B_ * T_, 3 * D_, D_);
  }

  // 3) per-head LN + RoPE -> head-major q,k,v bf16
  ln_rope<<<B_ * T_, 256, 0, stream>>>(qkvb, qsc, qbi, ksc, kbi, qo, ko, vo);

  // 4) causal flash attention -> attn bf16 [B,T,D]
  {
    dim3 g(T_ / 128, NH_, B_);
    flash_attn<<<g, 256, 0, stream>>>(qo, ko, vo, ab);
  }

  // 5) out = attn @ Wout^T -> fp32 into d_out
  {
    dim3 g(2048 / 128, 4096 / 128);
    gemm_bt<true><<<g, 256, 0, stream>>>(ab, wob, out, nullptr,
                                         B_ * T_, D_, D_);
  }
}
